// MipNeRF360MLP_31181462569336
// MI455X (gfx1250) — matrix operand order repaired and checked by reading, not compiled, not run on hardware
//
#include <hip/hip_runtime.h>
#include <hip/hip_bf16.h>
#include <math.h>

typedef _Float16 half_t;
typedef __attribute__((ext_vector_type(16))) _Float16 v16h;
typedef __attribute__((ext_vector_type(8)))  float    v8f;

// ---------------- WMMA helpers (CDNA5 16x16x32 f16 -> f32) ----------------

__device__ __forceinline__ v8f wmma16(v16h a, v16h b, v8f c) {
  // D = A(16x32 f16) * B(32x16 f16) + C(16x16 f32)
  return __builtin_amdgcn_wmma_f32_16x16x32_f16(false, a, false, b, (short)0, c,
                                                false, false);
}

__device__ __forceinline__ v8f zero8() {
  v8f z;
#pragma unroll
  for (int i = 0; i < 8; ++i) z[i] = 0.f;
  return z;
}

// A fragment: 16x32 (MxK) f16, row-major src with leading dim ld.
// lane l: m = l&15, koff = (l>>4)*16; elem i -> k = koff + i
__device__ __forceinline__ v16h load_a(const half_t* p, int ld, int lane) {
  const int m = lane & 15;
  const int koff = (lane >> 4) * 16;
  const half_t* row = p + (size_t)m * ld + koff;
  v16h a;
#pragma unroll
  for (int i = 0; i < 8; ++i) a[i] = row[i];
#pragma unroll
  for (int i = 0; i < 8; ++i) a[i + 8] = row[8 + i];
  return a;
}

// B fragment: 32x16 (KxN) f16 from row-major [K][N] src, leading dim ld.
// lane l: n = l&15, koff = (l>>4)*16; elem i -> k = koff + i
__device__ __forceinline__ v16h load_b(const half_t* p, int ld, int lane) {
  const int n = lane & 15;
  const int koff = (lane >> 4) * 16;
  v16h b;
#pragma unroll
  for (int i = 0; i < 16; ++i) b[i] = p[(size_t)(koff + i) * ld + n];
  return b;
}

// B fragment where B[k][n] = src[n][k] (i.e. src row-major [N][K], ld = row len)
__device__ __forceinline__ v16h load_bt(const half_t* p, int ld, int lane) {
  const int n = lane & 15;
  const int koff = (lane >> 4) * 16;
  const half_t* row = p + (size_t)n * ld + koff;
  v16h b;
#pragma unroll
  for (int i = 0; i < 16; ++i) b[i] = row[i];
  return b;
}

__device__ __forceinline__ float rsum16(float v) {
#pragma unroll
  for (int m = 1; m < 16; m <<= 1) v += __shfl_xor(v, m, 16);
  return v;
}
__device__ __forceinline__ float rmax16(float v) {
#pragma unroll
  for (int m = 1; m < 16; m <<= 1) v = fmaxf(v, __shfl_xor(v, m, 16));
  return v;
}

// per-row LN stats over a 16x64 C-layout tile (4 col tiles x 8 row elems)
__device__ __forceinline__ void ln_stats64(const float x[4][8], float* mn, float* rs) {
#pragma unroll
  for (int r = 0; r < 8; ++r) {
    float a = 0.f, q = 0.f;
#pragma unroll
    for (int t = 0; t < 4; ++t) { a += x[t][r]; q += x[t][r] * x[t][r]; }
    a = rsum16(a); q = rsum16(q);
    float m = a * (1.f / 64.f);
    float v = q * (1.f / 64.f) - m * m;
    mn[r] = m;
    rs[r] = rsqrtf(v + 1e-5f);
  }
}

// ---- CDNA5 async global->LDS copy (ASYNCcnt path, ISA 08_async_tensor §4) ----
// the LDS address is the LDS base plus the VDST value; flat LDS pointers carry the wave-relative
// LDS byte offset in the low 32 bits, which is exactly what VDST expects.
__device__ __forceinline__ void async_g2l_b128(void* lds_ptr, const void* gptr) {
  unsigned lofs = (unsigned)(uintptr_t)lds_ptr;
  unsigned long long ga = (unsigned long long)(uintptr_t)gptr;
  asm volatile("global_load_async_to_lds_b128 %0, %1, off"
               :
               : "v"(lofs), "v"(ga)
               : "memory");
}
__device__ __forceinline__ void wait_async0() {
  asm volatile("s_wait_asynccnt 0x0" ::: "memory");
}

// Copy 8192B (64x64 f16) global -> LDS using async-to-LDS, 256 threads.
__device__ __forceinline__ void async_copy_chunk(half_t* dst, const half_t* src,
                                                 int tid) {
#pragma unroll
  for (int i = 0; i < 2; ++i) {
    int o = (i * 256 + tid) * 16;  // bytes
    async_g2l_b128((char*)dst + o, (const char*)src + o);
  }
}

// ---------------- small utility kernels (stage 1, tiny cost) ----------------

__global__ void bcast_k(const float* __restrict__ in, float* __restrict__ out,
                        int per, int total) {
  int i = blockIdx.x * 256 + threadIdx.x;
  if (i < total) out[i] = in[i % per];
}

__global__ void ln_k(const float* __restrict__ x, const float* __restrict__ g,
                     const float* __restrict__ b, float* __restrict__ y, int dim) {
  int row = blockIdx.x, tid = threadIdx.x;
  const float* xr = x + (size_t)row * dim;
  __shared__ float s1[256], s2[256];
  float v = (tid < dim) ? xr[tid] : 0.f;
  s1[tid] = v; s2[tid] = v * v;
  __syncthreads();
  for (int s = 128; s > 0; s >>= 1) {
    if (tid < s) { s1[tid] += s1[tid + s]; s2[tid] += s2[tid + s]; }
    __syncthreads();
  }
  float mean = s1[0] / dim;
  float var = s2[0] / dim - mean * mean;
  float rstd = rsqrtf(var + 1e-5f);
  if (tid < dim) y[(size_t)row * dim + tid] = (v - mean) * rstd * g[tid] + b[tid];
}

// C[r,n] = sum_k A[r*as + k] * W[k*N + n] (+bias[n]) (+res[r*N+n])
__global__ void gemm_k(const float* __restrict__ A, const float* __restrict__ W,
                       const float* __restrict__ bias, const float* __restrict__ res,
                       float* __restrict__ C, int rows, int Kd, int N, int as) {
  int idx = blockIdx.x * 256 + threadIdx.x;
  if (idx >= rows * N) return;
  int r = idx / N, n = idx % N;
  const float* a = A + (size_t)r * as;
  float acc = 0.f;
  for (int k = 0; k < Kd; ++k) acc += a[k] * W[(size_t)k * N + n];
  if (bias) acc += bias[n];
  if (res) acc += res[(size_t)r * N + n];
  C[(size_t)r * N + n] = acc;
}

__global__ void geglu_k(const float* __restrict__ t, float* __restrict__ u,
                        int half, int total) {
  int i = blockIdx.x * 256 + threadIdx.x;
  if (i >= total) return;
  int r = i / half, c = i % half;
  float a = t[(size_t)r * 2 * half + c];
  float g = t[(size_t)r * 2 * half + half + c];
  float gl = 0.5f * g * (1.f + erff(g * 0.70710678118654752f));
  u[i] = a * gl;
}

// [B,R,H*D] f32 -> [B*H,R,D] f16 (with scale); input row stride = stride floats
__global__ void pack_k(const float* __restrict__ in, half_t* __restrict__ out,
                       int R, int H, int D, int stride, float scale, int total) {
  int i = blockIdx.x * 256 + threadIdx.x;
  if (i >= total) return;
  int d = i % D, h = (i / D) % H, r = (i / (D * H)) % R, b = i / (D * H * R);
  float v = in[((size_t)b * R + r) * stride + h * D + d] * scale;
  out[(((size_t)(b * H + h)) * R + r) * D + d] = (half_t)v;
}

// [B*H,R,D] f32 -> [B,R,H*D] f32
__global__ void unpack_k(const float* __restrict__ in, float* __restrict__ out,
                         int R, int H, int D, int total) {
  int i = blockIdx.x * 256 + threadIdx.x;
  if (i >= total) return;
  int d = i % D, h = (i / D) % H, r = (i / (D * H)) % R, b = i / (D * H * R);
  out[((size_t)b * R + r) * (H * D) + h * D + d] =
      in[(((size_t)(b * H + h)) * R + r) * D + d];
}

// lat[row,0:128] -> LN -> @ qa_wkv[128,128] -> K,V f16 [row,64] each
__global__ void kv2_k(const float* __restrict__ lat, const float* __restrict__ g,
                      const float* __restrict__ bb, const float* __restrict__ wkv,
                      half_t* __restrict__ Kg, half_t* __restrict__ Vg) {
  int row = blockIdx.x, tid = threadIdx.x;  // 128 threads
  __shared__ float cn[128], r1[128], r2[128];
  float v = lat[(size_t)row * 128 + tid];
  r1[tid] = v; r2[tid] = v * v;
  __syncthreads();
  for (int s = 64; s > 0; s >>= 1) {
    if (tid < s) { r1[tid] += r1[tid + s]; r2[tid] += r2[tid + s]; }
    __syncthreads();
  }
  float mean = r1[0] / 128.f;
  float var = r2[0] / 128.f - mean * mean;
  float rstd = rsqrtf(var + 1e-5f);
  cn[tid] = (v - mean) * rstd * g[tid] + bb[tid];
  __syncthreads();
  float acc = 0.f;
  for (int k = 0; k < 128; ++k) acc += cn[k] * wkv[(size_t)k * 128 + tid];
  if (tid < 64) Kg[(size_t)row * 64 + tid] = (half_t)acc;
  else          Vg[(size_t)row * 64 + (tid - 64)] = (half_t)acc;
}

// ---------------- generic WMMA flash attention, head dim 64 ----------------
// Q,K,V f16: [G,R,64]/[G,J,64]; Q pre-scaled by 1/8. O f32 [G,R,64].
// block = 8 waves; each wave owns 16 query rows; K/V chunked by 64 into LDS
// via CDNA5 async global->LDS loads.
__global__ void __launch_bounds__(256) flash64_k(const half_t* __restrict__ Qh,
                                                 const half_t* __restrict__ Kh,
                                                 const half_t* __restrict__ Vh,
                                                 float* __restrict__ O, int R, int J) {
  extern __shared__ char smemRaw[];
  half_t* sK = (half_t*)smemRaw;          // 64*64
  half_t* sV = sK + 4096;                 // 64*64
  half_t* sPbase = sV + 4096;             // 8 * 16*64
  const int tid = threadIdx.x, wave = tid >> 5, lane = tid & 31;
  const int nn = lane & 15, mb = (lane >> 4) * 8;
  const int g = blockIdx.y;
  const int row0 = blockIdx.x * 128 + wave * 16;
  half_t* sP = sPbase + wave * 1024;

  const half_t* Qg = Qh + ((size_t)g * R + row0) * 64;
  v16h qa0 = load_a(Qg, 64, lane);
  v16h qa1 = load_a(Qg + 32, 64, lane);

  v8f o[4];
#pragma unroll
  for (int t = 0; t < 4; ++t) o[t] = zero8();
  float mrow[8], lrow[8];
#pragma unroll
  for (int r = 0; r < 8; ++r) { mrow[r] = -1e30f; lrow[r] = 0.f; }

  for (int j0 = 0; j0 < J; j0 += 64) {
    __syncthreads();  // previous chunk fully consumed
    async_copy_chunk(sK, Kh + ((size_t)g * J + j0) * 64, tid);
    async_copy_chunk(sV, Vh + ((size_t)g * J + j0) * 64, tid);
    wait_async0();
    __syncthreads();  // chunk visible to all waves

    v8f s[4];
#pragma unroll
    for (int jt = 0; jt < 4; ++jt) {
      v16h kb0 = load_bt(sK + jt * 16 * 64, 64, lane);
      v16h kb1 = load_bt(sK + jt * 16 * 64 + 32, 64, lane);
      s[jt] = wmma16(qa1, kb1, wmma16(qa0, kb0, zero8()));
    }
    float mnew[8], al[8], cs[8];
#pragma unroll
    for (int r = 0; r < 8; ++r) {
      float cm = s[0][r];
#pragma unroll
      for (int jt = 1; jt < 4; ++jt) cm = fmaxf(cm, s[jt][r]);
      cm = rmax16(cm);
      mnew[r] = fmaxf(mrow[r], cm);
      al[r] = __expf(mrow[r] - mnew[r]);
      cs[r] = 0.f;
    }
#pragma unroll
    for (int jt = 0; jt < 4; ++jt)
#pragma unroll
      for (int r = 0; r < 8; ++r) {
        float p = __expf(s[jt][r] - mnew[r]);
        s[jt][r] = p;
        cs[r] += p;
      }
#pragma unroll
    for (int r = 0; r < 8; ++r) {
      cs[r] = rsum16(cs[r]);
      lrow[r] = lrow[r] * al[r] + cs[r];
      mrow[r] = mnew[r];
    }
#pragma unroll
    for (int t = 0; t < 4; ++t)
#pragma unroll
      for (int r = 0; r < 8; ++r) o[t][r] *= al[r];
#pragma unroll
    for (int jt = 0; jt < 4; ++jt)
#pragma unroll
      for (int r = 0; r < 8; ++r)
        sP[(mb + r) * 64 + jt * 16 + nn] = (half_t)s[jt][r];
    v16h pa0 = load_a(sP, 64, lane);
    v16h pa1 = load_a(sP + 32, 64, lane);
#pragma unroll
    for (int vt = 0; vt < 4; ++vt) {
      v16h vb0 = load_b(sV + vt * 16, 64, lane);
      v16h vb1 = load_b(sV + 32 * 64 + vt * 16, 64, lane);
      o[vt] = wmma16(pa1, vb1, wmma16(pa0, vb0, o[vt]));
    }
  }
#pragma unroll
  for (int t = 0; t < 4; ++t)
#pragma unroll
    for (int r = 0; r < 8; ++r)
      O[((size_t)g * R + row0 + mb + r) * 64 + t * 16 + nn] = o[t][r] / lrow[r];
}

// ---------------- fused stage-2 kernel (dominant cost, fully fused) --------
// Per 16-row query tile (one wave): LN -> Q=yn@wq (WMMA) -> flash attn over
// 512-latent K/V (async LDS chunks of 64) -> O@wo + bo + y -> LN -> GEGLU FF
// (WMMA) -> residual -> store. Single pass over the 64MB queries / 64MB out.
__global__ void __launch_bounds__(256) stage2_fused_k(
    const float* __restrict__ Yin, const half_t* __restrict__ Kg,
    const half_t* __restrict__ Vg, const float* __restrict__ lng,
    const float* __restrict__ lnb, const float* __restrict__ wq,
    const float* __restrict__ wo, const float* __restrict__ bo,
    const float* __restrict__ fng, const float* __restrict__ fnb,
    const float* __restrict__ w1, const float* __restrict__ fb1,
    const float* __restrict__ w2, const float* __restrict__ fb2,
    float* __restrict__ Yout) {
  extern __shared__ char smemRaw[];
  half_t* sWq = (half_t*)smemRaw;  // 64*64
  half_t* sWo = sWq + 4096;        // 64*64
  half_t* sW1 = sWo + 4096;        // 64*128
  half_t* sW2 = sW1 + 8192;        // 64*64
  half_t* sK = sW2 + 4096;         // 64*64 chunk
  half_t* sV = sK + 4096;          // 64*64 chunk
  half_t* sScr = sV + 4096;        // 8 waves * 16*64

  const int tid = threadIdx.x, wave = tid >> 5, lane = tid & 31;
  const int nn = lane & 15, mb = (lane >> 4) * 8;
  const int b = blockIdx.y;
  const int row0 = blockIdx.x * 128 + wave * 16;
  half_t* sP = sScr + wave * 1024;

  for (int i = tid; i < 4096; i += 256) {
    sWq[i] = (half_t)wq[i];
    sWo[i] = (half_t)wo[i];
    sW2[i] = (half_t)w2[i];
  }
  for (int i = tid; i < 8192; i += 256) sW1[i] = (half_t)w1[i];

  float gA[4], bA[4], gF[4], bF[4], boL[4], b2L[4], b1L[8];
#pragma unroll
  for (int t = 0; t < 4; ++t) {
    int c = t * 16 + nn;
    gA[t] = lng[c]; bA[t] = lnb[c];
    gF[t] = fng[c]; bF[t] = fnb[c];
    boL[t] = bo[c]; b2L[t] = fb2[c];
  }
#pragma unroll
  for (int t = 0; t < 8; ++t) b1L[t] = fb1[t * 16 + nn];
  __syncthreads();

  // load y tile and LN
  const float* Yrow = Yin + ((size_t)b * 65536 + row0) * 64;
  float y[4][8];
#pragma unroll
  for (int t = 0; t < 4; ++t)
#pragma unroll
    for (int r = 0; r < 8; ++r) y[t][r] = Yrow[(mb + r) * 64 + t * 16 + nn];
  float mn[8], rs[8];
  ln_stats64(y, mn, rs);
#pragma unroll
  for (int t = 0; t < 4; ++t)
#pragma unroll
    for (int r = 0; r < 8; ++r)
      sP[(mb + r) * 64 + t * 16 + nn] =
          (half_t)((y[t][r] - mn[r]) * rs[r] * gA[t] + bA[t]);

  // Q = yn @ wq, scaled by 1/sqrt(64)
  v16h a0 = load_a(sP, 64, lane), a1 = load_a(sP + 32, 64, lane);
  v8f qc[4];
#pragma unroll
  for (int nt = 0; nt < 4; ++nt) {
    v16h w0v = load_b(sWq + nt * 16, 64, lane);
    v16h w1v = load_b(sWq + 32 * 64 + nt * 16, 64, lane);
    qc[nt] = wmma16(a1, w1v, wmma16(a0, w0v, zero8()));
  }
#pragma unroll
  for (int nt = 0; nt < 4; ++nt)
#pragma unroll
    for (int r = 0; r < 8; ++r)
      sP[(mb + r) * 64 + nt * 16 + nn] = (half_t)(qc[nt][r] * 0.125f);
  v16h qa0 = load_a(sP, 64, lane), qa1 = load_a(sP + 32, 64, lane);

  // flash attention over 512 context rows, chunks of 64
  v8f o[4];
#pragma unroll
  for (int t = 0; t < 4; ++t) o[t] = zero8();
  float mrow[8], lrow[8];
#pragma unroll
  for (int r = 0; r < 8; ++r) { mrow[r] = -1e30f; lrow[r] = 0.f; }

  for (int j0 = 0; j0 < 512; j0 += 64) {
    __syncthreads();  // previous chunk fully consumed
    async_copy_chunk(sK, Kg + ((size_t)b * 512 + j0) * 64, tid);
    async_copy_chunk(sV, Vg + ((size_t)b * 512 + j0) * 64, tid);
    wait_async0();
    __syncthreads();  // chunk visible to all waves

    v8f s[4];
#pragma unroll
    for (int jt = 0; jt < 4; ++jt) {
      v16h kb0 = load_bt(sK + jt * 16 * 64, 64, lane);
      v16h kb1 = load_bt(sK + jt * 16 * 64 + 32, 64, lane);
      s[jt] = wmma16(qa1, kb1, wmma16(qa0, kb0, zero8()));
    }
    float mnew[8], al[8], cs[8];
#pragma unroll
    for (int r = 0; r < 8; ++r) {
      float cm = s[0][r];
#pragma unroll
      for (int jt = 1; jt < 4; ++jt) cm = fmaxf(cm, s[jt][r]);
      cm = rmax16(cm);
      mnew[r] = fmaxf(mrow[r], cm);
      al[r] = __expf(mrow[r] - mnew[r]);
      cs[r] = 0.f;
    }
#pragma unroll
    for (int jt = 0; jt < 4; ++jt)
#pragma unroll
      for (int r = 0; r < 8; ++r) {
        float p = __expf(s[jt][r] - mnew[r]);
        s[jt][r] = p;
        cs[r] += p;
      }
#pragma unroll
    for (int r = 0; r < 8; ++r) {
      cs[r] = rsum16(cs[r]);
      lrow[r] = lrow[r] * al[r] + cs[r];
      mrow[r] = mnew[r];
    }
#pragma unroll
    for (int t = 0; t < 4; ++t)
#pragma unroll
      for (int r = 0; r < 8; ++r) o[t][r] *= al[r];
#pragma unroll
    for (int jt = 0; jt < 4; ++jt)
#pragma unroll
      for (int r = 0; r < 8; ++r)
        sP[(mb + r) * 64 + jt * 16 + nn] = (half_t)s[jt][r];
    v16h pa0 = load_a(sP, 64, lane), pa1 = load_a(sP + 32, 64, lane);
#pragma unroll
    for (int vt = 0; vt < 4; ++vt) {
      v16h vb0 = load_b(sV + vt * 16, 64, lane);
      v16h vb1 = load_b(sV + 32 * 64 + vt * 16, 64, lane);
      o[vt] = wmma16(pa1, vb1, wmma16(pa0, vb0, o[vt]));
    }
  }

  // out projection + bias + residual
#pragma unroll
  for (int t = 0; t < 4; ++t)
#pragma unroll
    for (int r = 0; r < 8; ++r)
      sP[(mb + r) * 64 + t * 16 + nn] = (half_t)(o[t][r] / lrow[r]);
  v16h oa0 = load_a(sP, 64, lane), oa1 = load_a(sP + 32, 64, lane);
  float y2[4][8];
#pragma unroll
  for (int nt = 0; nt < 4; ++nt) {
    v16h w0v = load_b(sWo + nt * 16, 64, lane);
    v16h w1v = load_b(sWo + 32 * 64 + nt * 16, 64, lane);
    v8f c = wmma16(oa1, w1v, wmma16(oa0, w0v, zero8()));
#pragma unroll
    for (int r = 0; r < 8; ++r) y2[nt][r] = c[r] + boL[nt] + y[nt][r];
  }

  // FF: LN -> h@w1 + b1 -> GEGLU -> @w2 + b2 -> residual -> store
  ln_stats64(y2, mn, rs);
#pragma unroll
  for (int t = 0; t < 4; ++t)
#pragma unroll
    for (int r = 0; r < 8; ++r)
      sP[(mb + r) * 64 + t * 16 + nn] =
          (half_t)((y2[t][r] - mn[r]) * rs[r] * gF[t] + bF[t]);
  v16h ha0 = load_a(sP, 64, lane), ha1 = load_a(sP + 32, 64, lane);
  v8f tt[8];
#pragma unroll
  for (int nt = 0; nt < 8; ++nt) {
    v16h w0v = load_b(sW1 + nt * 16, 128, lane);
    v16h w1v = load_b(sW1 + 32 * 128 + nt * 16, 128, lane);
    tt[nt] = wmma16(ha1, w1v, wmma16(ha0, w0v, zero8()));
  }
#pragma unroll
  for (int t = 0; t < 4; ++t)
#pragma unroll
    for (int r = 0; r < 8; ++r) {
      float av = tt[t][r] + b1L[t];
      float gv = tt[t + 4][r] + b1L[t + 4];
      float gl = 0.5f * gv * (1.f + erff(gv * 0.70710678118654752f));
      sP[(mb + r) * 64 + t * 16 + nn] = (half_t)(av * gl);
    }
  v16h ua0 = load_a(sP, 64, lane), ua1 = load_a(sP + 32, 64, lane);
  float* Orow = Yout + ((size_t)b * 65536 + row0) * 64;
#pragma unroll
  for (int nt = 0; nt < 4; ++nt) {
    v16h w0v = load_b(sW2 + nt * 16, 64, lane);
    v16h w1v = load_b(sW2 + 32 * 64 + nt * 16, 64, lane);
    v8f c = wmma16(ua1, w1v, wmma16(ua0, w0v, zero8()));
#pragma unroll
    for (int r = 0; r < 8; ++r)
      Orow[(mb + r) * 64 + nt * 16 + nn] = c[r] + b2L[nt] + y2[nt][r];
  }
}

// ---------------------------------------------------------------------------

extern "C" void kernel_launch(void* const* d_in, const int* in_sizes, int n_in,
                              void* d_out, int out_size, void* d_ws, size_t ws_size,
                              hipStream_t stream) {
  (void)in_sizes; (void)n_in; (void)out_size; (void)ws_size;
  const float* codebook = (const float*)d_in[0];
  const float* queries  = (const float*)d_in[1];
  const float* latents  = (const float*)d_in[2];
  const float* ca_ng = (const float*)d_in[3],  *ca_nb = (const float*)d_in[4];
  const float* ca_cng = (const float*)d_in[5], *ca_cnb = (const float*)d_in[6];
  const float* ca_wq = (const float*)d_in[7],  *ca_wkv = (const float*)d_in[8];
  const float* ca_wo = (const float*)d_in[9],  *ca_bo = (const float*)d_in[10];
  const float* cf_ng = (const float*)d_in[11], *cf_nb = (const float*)d_in[12];
  const float* cf_w1 = (const float*)d_in[13], *cf_b1 = (const float*)d_in[14];
  const float* cf_w2 = (const float*)d_in[15], *cf_b2 = (const float*)d_in[16];
  const float* sa_ng = (const float*)d_in[17], *sa_nb = (const float*)d_in[18];
  const float* sa_wq = (const float*)d_in[19], *sa_wkv = (const float*)d_in[20];
  const float* sa_wo = (const float*)d_in[21], *sa_bo = (const float*)d_in[22];
  const float* sf_ng = (const float*)d_in[23], *sf_nb = (const float*)d_in[24];
  const float* sf_w1 = (const float*)d_in[25], *sf_b1 = (const float*)d_in[26];
  const float* sf_w2 = (const float*)d_in[27], *sf_b2 = (const float*)d_in[28];
  const float* qa_ng = (const float*)d_in[29], *qa_nb = (const float*)d_in[30];
  const float* qa_cng = (const float*)d_in[31], *qa_cnb = (const float*)d_in[32];
  const float* qa_wq = (const float*)d_in[33], *qa_wkv = (const float*)d_in[34];
  const float* qa_wo = (const float*)d_in[35], *qa_bo = (const float*)d_in[36];
  const float* qf_ng = (const float*)d_in[37], *qf_nb = (const float*)d_in[38];
  const float* qf_w1 = (const float*)d_in[39], *qf_b1 = (const float*)d_in[40];
  const float* qf_w2 = (const float*)d_in[41], *qf_b2 = (const float*)d_in[42];
  float* out = (float*)d_out;

  // workspace carve
  char* ws = (char*)d_ws;
  size_t off = 0;
  auto take = [&](size_t n) -> char* {
    char* p = ws + off;
    off += (n + 255) & ~(size_t)255;
    return p;
  };
  float* X0 = (float*)take((size_t)2048 * 128 * 4);
  float* XN = (float*)take((size_t)2048 * 128 * 4);
  float* CNb = (float*)take((size_t)4096 * 256 * 4);
  float* QF = (float*)take((size_t)2048 * 512 * 4);
  float* KVF = (float*)take((size_t)2048 * 1024 * 4);
  float* OF = (float*)take((size_t)32 * 512 * 64 * 4);
  float* OU = (float*)take((size_t)2048 * 512 * 4);
  float* XA = (float*)take((size_t)2048 * 128 * 4);
  float* XB = (float*)take((size_t)2048 * 128 * 4);
  float* TT = (float*)take((size_t)2048 * 1024 * 4);
  float* UU = (float*)take((size_t)2048 * 512 * 4);
  half_t* QH = (half_t*)take((size_t)32 * 512 * 64 * 2);
  half_t* KH = (half_t*)take((size_t)32 * 512 * 64 * 2);
  half_t* VH = (half_t*)take((size_t)32 * 512 * 64 * 2);
  half_t* K2 = (half_t*)take((size_t)4 * 512 * 64 * 2);
  half_t* V2 = (half_t*)take((size_t)4 * 512 * 64 * 2);

  const int flashSmem = (4096 + 4096 + 8192) * 2;                    // 32 KB
  const int megaSmem = (4096 * 5 + 8192 + 8192) * 2;                 // 72 KB

  // ---- Stage 1: CodebookAttention ----
  bcast_k<<<1024, 256, 0, stream>>>(latents, X0, 512 * 128, 2048 * 128);
  ln_k<<<2048, 256, 0, stream>>>(X0, ca_ng, ca_nb, XN, 128);
  ln_k<<<4096, 256, 0, stream>>>(codebook, ca_cng, ca_cnb, CNb, 256);
  gemm_k<<<(2048 * 64) / 256, 256, 0, stream>>>(XN, ca_wq, nullptr, nullptr, QF, 2048, 128, 64, 128);
  gemm_k<<<(4096 * 128) / 256, 256, 0, stream>>>(CNb, ca_wkv, nullptr, nullptr, KVF, 4096, 256, 128, 256);
  pack_k<<<(4 * 512 * 64) / 256, 256, 0, stream>>>(QF, QH, 512, 1, 64, 64, 0.125f, 4 * 512 * 64);
  pack_k<<<(4 * 1024 * 64) / 256, 256, 0, stream>>>(KVF, KH, 1024, 1, 64, 128, 1.f, 4 * 1024 * 64);
  pack_k<<<(4 * 1024 * 64) / 256, 256, 0, stream>>>(KVF + 64, VH, 1024, 1, 64, 128, 1.f, 4 * 1024 * 64);
  flash64_k<<<dim3(4, 4), 256, flashSmem, stream>>>(QH, KH, VH, OF, 512, 1024);
  unpack_k<<<(4 * 512 * 64) / 256, 256, 0, stream>>>(OF, OU, 512, 1, 64, 4 * 512 * 64);
  gemm_k<<<(2048 * 128) / 256, 256, 0, stream>>>(OU, ca_wo, ca_bo, X0, XA, 2048, 64, 128, 64);
  ln_k<<<2048, 256, 0, stream>>>(XA, cf_ng, cf_nb, XN, 128);
  gemm_k<<<(2048 * 1024) / 256, 256, 0, stream>>>(XN, cf_w1, cf_b1, nullptr, TT, 2048, 128, 1024, 128);
  geglu_k<<<(2048 * 512) / 256, 256, 0, stream>>>(TT, UU, 512, 2048 * 512);
  gemm_k<<<(2048 * 128) / 256, 256, 0, stream>>>(UU, cf_w2, cf_b2, XA, XB, 2048, 512, 128, 512);
  // latent self-attention (8 heads x 64)
  ln_k<<<2048, 256, 0, stream>>>(XB, sa_ng, sa_nb, XN, 128);
  gemm_k<<<(2048 * 512) / 256, 256, 0, stream>>>(XN, sa_wq, nullptr, nullptr, QF, 2048, 128, 512, 128);
  gemm_k<<<(2048 * 1024) / 256, 256, 0, stream>>>(XN, sa_wkv, nullptr, nullptr, KVF, 2048, 128, 1024, 128);
  pack_k<<<(4 * 512 * 512) / 256, 256, 0, stream>>>(QF, QH, 512, 8, 64, 512, 0.125f, 4 * 512 * 512);
  pack_k<<<(4 * 512 * 512) / 256, 256, 0, stream>>>(KVF, KH, 512, 8, 64, 1024, 1.f, 4 * 512 * 512);
  pack_k<<<(4 * 512 * 512) / 256, 256, 0, stream>>>(KVF + 512, VH, 512, 8, 64, 1024, 1.f, 4 * 512 * 512);
  flash64_k<<<dim3(4, 32), 256, flashSmem, stream>>>(QH, KH, VH, OF, 512, 512);
  unpack_k<<<(4 * 512 * 512) / 256, 256, 0, stream>>>(OF, OU, 512, 8, 64, 4 * 512 * 512);
  gemm_k<<<(2048 * 128) / 256, 256, 0, stream>>>(OU, sa_wo, sa_bo, XB, XA, 2048, 512, 128, 512);
  ln_k<<<2048, 256, 0, stream>>>(XA, sf_ng, sf_nb, XN, 128);
  gemm_k<<<(2048 * 1024) / 256, 256, 0, stream>>>(XN, sf_w1, sf_b1, nullptr, TT, 2048, 128, 1024, 128);
  geglu_k<<<(2048 * 512) / 256, 256, 0, stream>>>(TT, UU, 512, 2048 * 512);
  gemm_k<<<(2048 * 128) / 256, 256, 0, stream>>>(UU, sf_w2, sf_b2, XA, XB, 2048, 512, 128, 512);  // lat

  // ---- Stage 2: CoordinateAttention (fused, dominant) ----
  kv2_k<<<2048, 128, 0, stream>>>(XB, qa_cng, qa_cnb, qa_wkv, K2, V2);
  stage2_fused_k<<<dim3(512, 4), 256, megaSmem, stream>>>(
      queries, K2, V2, qa_ng, qa_nb, qa_wq, qa_wo, qa_bo, qf_ng, qf_nb, qf_w1,
      qf_b1, qf_w2, qf_b2, out);
}
